// MultiHeadAttn_39754217292599
// MI455X (gfx1250) — matrix operand order repaired and checked by reading, not compiled, not run on hardware
//
#include <hip/hip_runtime.h>

#define D_MODEL 1024
#define HEADS   16
#define DKH     64
#define BATCH   4
#define SEQ     2048
#define N_TOK   (BATCH * SEQ)

#define USE_ASYNC_LDS 1

typedef __attribute__((ext_vector_type(16))) _Float16 v16h;
typedef __attribute__((ext_vector_type(8)))  _Float16 v8h;
typedef __attribute__((ext_vector_type(4)))  _Float16 v4h;
typedef __attribute__((ext_vector_type(8)))  float    v8f;
typedef __attribute__((ext_vector_type(4)))  float    v4f;

__device__ __forceinline__ v8f wmma_f16(v16h a, v16h b, v8f c) {
  // D = A(16x32 f16) * B(32x16 f16) + C(16x16 f32)
  return __builtin_amdgcn_wmma_f32_16x16x32_f16(false, a, false, b, (short)0, c,
                                                false, false);
}

__device__ __forceinline__ float lane_bcast(float v, int srcLane) {
  return __int_as_float(__builtin_amdgcn_ds_bpermute(srcLane << 2, __float_as_int(v)));
}

// ---- gfx1250 async global->LDS copy (16 bytes per lane) ---------------------
__device__ __forceinline__ void async_cp16(void* lds_dst, const void* gsrc) {
#if USE_ASYNC_LDS
  unsigned loff = (unsigned)(unsigned long long)lds_dst;   // low 32 bits = LDS offset
  unsigned long long ga = (unsigned long long)gsrc;
  asm volatile("global_load_async_to_lds_b128 %0, %1, off"
               :: "v"(loff), "v"(ga)
               : "memory");
#else
  *(v8h*)lds_dst = *(const v8h*)gsrc;
#endif
}

__device__ __forceinline__ void wait_async() {
#if USE_ASYNC_LDS
#if __has_builtin(__builtin_amdgcn_s_wait_asynccnt)
  __builtin_amdgcn_s_wait_asynccnt(0);
#else
  asm volatile("s_wait_asynccnt 0" ::: "memory");
#endif
#endif
}

// A operand (16-bit): lane holds row m = lane&15; elements 0..15 = K k+2*c0..+15
// (2*c0 = 0 lanes 0-15, 16 lanes 16-31). f32 -> f16.
__device__ __forceinline__ v16h cvtA_f32(const float* row, int k, int c0) {
  const v4f* p0 = (const v4f*)(row + k + 2 * c0);
  const v4f* p1 = (const v4f*)(row + k + 2 * c0 + 8);
  v4f a0 = p0[0], a1 = p0[1], b0 = p1[0], b1 = p1[1];
  v16h r;
#pragma unroll
  for (int i = 0; i < 4; ++i) {
    r[i]      = (_Float16)a0[i];
    r[4 + i]  = (_Float16)a1[i];
    r[8 + i]  = (_Float16)b0[i];
    r[12 + i] = (_Float16)b1[i];
  }
  return r;
}

// A operand already f16 (works for global or LDS pointers)
__device__ __forceinline__ v16h ldA_f16(const _Float16* row, int k, int c0) {
  v8h lo = *(const v8h*)(row + k + 2 * c0);
  v8h hi = *(const v8h*)(row + k + 2 * c0 + 8);
  v16h r;
#pragma unroll
  for (int i = 0; i < 8; ++i) { r[i] = lo[i]; r[8 + i] = hi[i]; }
  return r;
}

// ---------------- weight pre-conversion: f32 -> f16 (row-major) --------------
__global__ __launch_bounds__(256) void cvt_kernel(const float* __restrict__ s,
                                                  _Float16* __restrict__ d) {
  const size_t i = (size_t)(blockIdx.x * 256 + threadIdx.x) * 4;
  v4f x = *(const v4f*)(s + i);
  v4h r;
#pragma unroll
  for (int j = 0; j < 4; ++j) r[j] = (_Float16)x[j];
  *(v4h*)(d + i) = r;
}

// ---------------- QKV projection: P = X @ W^T + bias (32x64 per wave) --------
// VLAYOUT 0: out[((b*H+h)*SEQ + s)*64 + dk]   (Q, K)
// VLAYOUT 1: out[((b*H+h)*64 + dk)*SEQ + s]   (V transposed for P·V B-operand)
template <int VLAYOUT>
__global__ __launch_bounds__(256) void proj_kernel(
    const float* __restrict__ X, const _Float16* __restrict__ Wh,
    const float* __restrict__ bias, _Float16* __restrict__ out) {
  const int lane = threadIdx.x & 31;
  const int wave = threadIdx.x >> 5;
  const int t = blockIdx.x * 8 + wave;           // 256 x 16 macro tiles
  const int n0 = (t >> 4) * 32;
  const int j0 = (t & 15) * 64;
  const int lrow = lane & 15, hi = lane >> 4, c0 = hi * 8, klo = hi * 16;

  const float* xr0 = X + (size_t)(n0 + lrow) * D_MODEL;
  const float* xr1 = X + (size_t)(n0 + 16 + lrow) * D_MODEL;
  const _Float16* wr0 = Wh + (size_t)(j0 + 0 * 16 + lrow) * D_MODEL;
  const _Float16* wr1 = Wh + (size_t)(j0 + 1 * 16 + lrow) * D_MODEL;
  const _Float16* wr2 = Wh + (size_t)(j0 + 2 * 16 + lrow) * D_MODEL;
  const _Float16* wr3 = Wh + (size_t)(j0 + 3 * 16 + lrow) * D_MODEL;

  v8f acc[2][4] = {};
  for (int k = 0; k < D_MODEL; k += 32) {
    v16h a0 = cvtA_f32(xr0, k, c0);
    v16h a1 = cvtA_f32(xr1, k, c0);
    v16h b0 = *(const v16h*)(wr0 + k + klo);
    v16h b1 = *(const v16h*)(wr1 + k + klo);
    v16h b2 = *(const v16h*)(wr2 + k + klo);
    v16h b3 = *(const v16h*)(wr3 + k + klo);
    acc[0][0] = wmma_f16(a0, b0, acc[0][0]);
    acc[1][0] = wmma_f16(a1, b0, acc[1][0]);
    acc[0][1] = wmma_f16(a0, b1, acc[0][1]);
    acc[1][1] = wmma_f16(a1, b1, acc[1][1]);
    acc[0][2] = wmma_f16(a0, b2, acc[0][2]);
    acc[1][2] = wmma_f16(a1, b2, acc[1][2]);
    acc[0][3] = wmma_f16(a0, b3, acc[0][3]);
    acc[1][3] = wmma_f16(a1, b3, acc[1][3]);
  }
#pragma unroll
  for (int jt = 0; jt < 4; ++jt) {
    const int j = j0 + jt * 16 + lrow;
    const float bj = bias[j];
    const int h = j >> 6, dk = j & 63;
#pragma unroll
    for (int mt = 0; mt < 2; ++mt) {
#pragma unroll
      for (int i = 0; i < 8; ++i) {
        const int n = n0 + mt * 16 + i + hi * 8;
        const int b_ = n >> 11;            // n / SEQ
        const int s = n & (SEQ - 1);
        const float v = acc[mt][jt][i] + bj;
        size_t idx;
        if (VLAYOUT)
          idx = ((size_t)(b_ * HEADS + h) * DKH + dk) * SEQ + s;
        else
          idx = ((size_t)(b_ * HEADS + h) * SEQ + s) * DKH + dk;
        out[idx] = (_Float16)v;
      }
    }
  }
}

// ---------------- Flash attention: block = 8 waves x 16 queries, one head ----
// K/V chunks staged in LDS via async copies, double-buffered.
__global__ __launch_bounds__(256) void attn_kernel(
    const _Float16* __restrict__ Qh, const _Float16* __restrict__ Kh,
    const _Float16* __restrict__ Vt, _Float16* __restrict__ X2) {
  __shared__ _Float16 lK[2][32 * 64];   // 32 keys x 64 dk
  __shared__ _Float16 lV[2][64 * 32];   // 64 dk x 32 keys (Vt layout)

  const int tid = threadIdx.x;
  const int lane = tid & 31;
  const int wave = tid >> 5;
  const int bh = blockIdx.x >> 4;                  // 16 blocks per (b,h)
  const int q0 = (blockIdx.x & 15) * 128 + wave * 16;
  const int b = bh >> 4, h = bh & 15;
  const int lrow = lane & 15, hi = lane >> 4, c0 = hi * 8, klo = hi * 16;

  const _Float16* Qp = Qh + (size_t)bh * SEQ * DKH;
  const _Float16* Kp = Kh + (size_t)bh * SEQ * DKH;
  const _Float16* Vp = Vt + (size_t)bh * DKH * SEQ;

  // cooperative staging: each thread moves one 16B packet per buffer
  const int krow = tid >> 3, kcol = (tid & 7) * 8;   // K chunk 32x64
  const int vrow = tid >> 2, vcol = (tid & 3) * 8;   // V chunk 64x32

  // prologue: stage chunk kb=0 into buffer 0
  async_cp16(&lK[0][krow * 64 + kcol], Kp + (size_t)krow * DKH + kcol);
  async_cp16(&lV[0][vrow * 32 + vcol], Vp + (size_t)vrow * SEQ + vcol);

  // Q as B-operand (lane n = query, K = dk), reused for all key chunks
  const _Float16* qrowp = Qp + (size_t)(q0 + lrow) * DKH;
  v16h bq0 = *(const v16h*)(qrowp + klo);
  v16h bq1 = *(const v16h*)(qrowp + 32 + klo);

  v8f O0 = {}, O1 = {}, O2 = {}, O3 = {};
  float m = -1e30f, ssum = 0.f;

  for (int kb = 0; kb < SEQ; kb += 32) {
    const int cur = (kb >> 5) & 1;
    wait_async();          // our staging loads for buffer `cur` are done
    __syncthreads();       // everyone's are done; prev buffer free to reuse
    if (kb + 32 < SEQ) {
      const int nxt = cur ^ 1;
      async_cp16(&lK[nxt][krow * 64 + kcol],
                 Kp + (size_t)(kb + 32 + krow) * DKH + kcol);
      async_cp16(&lV[nxt][vrow * 32 + vcol],
                 Vp + (size_t)vrow * SEQ + (kb + 32) + vcol);
    }

    // transposed score tiles St[key][query] from LDS
    const _Float16* k0r = &lK[cur][lrow * 64];
    const _Float16* k1r = &lK[cur][(16 + lrow) * 64];
    v8f st0 = {}, st1 = {};
    st0 = wmma_f16(ldA_f16(k0r, 0, c0), bq0, st0);
    st0 = wmma_f16(ldA_f16(k0r, 32, c0), bq1, st0);
    st1 = wmma_f16(ldA_f16(k1r, 0, c0), bq0, st1);
    st1 = wmma_f16(ldA_f16(k1r, 32, c0), bq1, st1);

    // online softmax; lane owns query q = lane&15 (keys split with lane^16)
    float cm = -1e30f;
#pragma unroll
    for (int i = 0; i < 8; ++i) {
      st0[i] *= 0.125f;   // 1/sqrt(64)
      st1[i] *= 0.125f;
      cm = fmaxf(cm, fmaxf(st0[i], st1[i]));
    }
    cm = fmaxf(cm, lane_bcast(cm, lane ^ 16));
    const float m_new = fmaxf(m, cm);
    const float alpha = __expf(m - m_new);

    v16h ap;   // probabilities packed directly in PV A-operand layout
    float psum = 0.f;
#pragma unroll
    for (int i = 0; i < 8; ++i) {
      const float p0 = __expf(st0[i] - m_new);
      const float p1 = __expf(st1[i] - m_new);
      psum += p0 + p1;
      ap[i] = (_Float16)p0;
      ap[8 + i] = (_Float16)p1;
    }
    psum += lane_bcast(psum, lane ^ 16);
    ssum = ssum * alpha + psum;
    m = m_new;

    float f[8];
#pragma unroll
    for (int i = 0; i < 8; ++i) f[i] = lane_bcast(alpha, i + hi * 8);
#pragma unroll
    for (int i = 0; i < 8; ++i) {
      O0[i] *= f[i]; O1[i] *= f[i]; O2[i] *= f[i]; O3[i] *= f[i];
    }

    // O += P(16x32) @ V(32x64); B-operand rows from LDS Vt chunk
    O0 = wmma_f16(ap, __builtin_shufflevector(*(const v8h*)(&lV[cur][(0 * 16 + lrow) * 32 + c0]), *(const v8h*)(&lV[cur][(0 * 16 + lrow) * 32 + 16 + c0]), 0, 1, 2, 3, 4, 5, 6, 7, 8, 9, 10, 11, 12, 13, 14, 15), O0);
    O1 = wmma_f16(ap, __builtin_shufflevector(*(const v8h*)(&lV[cur][(1 * 16 + lrow) * 32 + c0]), *(const v8h*)(&lV[cur][(1 * 16 + lrow) * 32 + 16 + c0]), 0, 1, 2, 3, 4, 5, 6, 7, 8, 9, 10, 11, 12, 13, 14, 15), O1);
    O2 = wmma_f16(ap, __builtin_shufflevector(*(const v8h*)(&lV[cur][(2 * 16 + lrow) * 32 + c0]), *(const v8h*)(&lV[cur][(2 * 16 + lrow) * 32 + 16 + c0]), 0, 1, 2, 3, 4, 5, 6, 7, 8, 9, 10, 11, 12, 13, 14, 15), O2);
    O3 = wmma_f16(ap, __builtin_shufflevector(*(const v8h*)(&lV[cur][(3 * 16 + lrow) * 32 + c0]), *(const v8h*)(&lV[cur][(3 * 16 + lrow) * 32 + 16 + c0]), 0, 1, 2, 3, 4, 5, 6, 7, 8, 9, 10, 11, 12, 13, 14, 15), O3);
  }

  const float inv = 1.f / ssum;
  float f[8];
#pragma unroll
  for (int i = 0; i < 8; ++i) f[i] = lane_bcast(inv, i + hi * 8);
#pragma unroll
  for (int i = 0; i < 8; ++i) {
    const int qrow = q0 + i + hi * 8;
    _Float16* dst = X2 + (size_t)(b * SEQ + qrow) * D_MODEL + h * DKH + lrow;
    dst[0]  = (_Float16)(O0[i] * f[i]);
    dst[16] = (_Float16)(O1[i] * f[i]);
    dst[32] = (_Float16)(O2[i] * f[i]);
    dst[48] = (_Float16)(O3[i] * f[i]);
  }
}

// ---------------- Output projection: Y = X2 @ Wo^T + bo (32x64 per wave) -----
__global__ __launch_bounds__(256) void oproj_kernel(
    const _Float16* __restrict__ X2, const _Float16* __restrict__ Woh,
    const float* __restrict__ bo, float* __restrict__ Y) {
  const int lane = threadIdx.x & 31;
  const int wave = threadIdx.x >> 5;
  const int t = blockIdx.x * 8 + wave;
  const int n0 = (t >> 4) * 32;
  const int j0 = (t & 15) * 64;
  const int lrow = lane & 15, hi = lane >> 4, c0 = hi * 8, klo = hi * 16;

  const _Float16* xr0 = X2 + (size_t)(n0 + lrow) * D_MODEL;
  const _Float16* xr1 = X2 + (size_t)(n0 + 16 + lrow) * D_MODEL;
  const _Float16* wr0 = Woh + (size_t)(j0 + 0 * 16 + lrow) * D_MODEL;
  const _Float16* wr1 = Woh + (size_t)(j0 + 1 * 16 + lrow) * D_MODEL;
  const _Float16* wr2 = Woh + (size_t)(j0 + 2 * 16 + lrow) * D_MODEL;
  const _Float16* wr3 = Woh + (size_t)(j0 + 3 * 16 + lrow) * D_MODEL;

  v8f acc[2][4] = {};
  for (int k = 0; k < D_MODEL; k += 32) {
    v16h a0 = ldA_f16(xr0, k, c0);
    v16h a1 = ldA_f16(xr1, k, c0);
    v16h b0 = *(const v16h*)(wr0 + k + klo);
    v16h b1 = *(const v16h*)(wr1 + k + klo);
    v16h b2 = *(const v16h*)(wr2 + k + klo);
    v16h b3 = *(const v16h*)(wr3 + k + klo);
    acc[0][0] = wmma_f16(a0, b0, acc[0][0]);
    acc[1][0] = wmma_f16(a1, b0, acc[1][0]);
    acc[0][1] = wmma_f16(a0, b1, acc[0][1]);
    acc[1][1] = wmma_f16(a1, b1, acc[1][1]);
    acc[0][2] = wmma_f16(a0, b2, acc[0][2]);
    acc[1][2] = wmma_f16(a1, b2, acc[1][2]);
    acc[0][3] = wmma_f16(a0, b3, acc[0][3]);
    acc[1][3] = wmma_f16(a1, b3, acc[1][3]);
  }
#pragma unroll
  for (int jt = 0; jt < 4; ++jt) {
    const float bj = bo[j0 + jt * 16 + lrow];
#pragma unroll
    for (int mt = 0; mt < 2; ++mt) {
#pragma unroll
      for (int i = 0; i < 8; ++i) {
        const int n = n0 + mt * 16 + i + hi * 8;
        Y[(size_t)n * D_MODEL + (j0 + jt * 16 + lrow)] = acc[mt][jt][i] + bj;
      }
    }
  }
}

extern "C" void kernel_launch(void* const* d_in, const int* in_sizes, int n_in,
                              void* d_out, int out_size, void* d_ws, size_t ws_size,
                              hipStream_t stream) {
  const float* q  = (const float*)d_in[0];
  const float* k  = (const float*)d_in[1];
  const float* v  = (const float*)d_in[2];
  // d_in[3] = mask (bool) — has no effect in the reference, faithfully ignored.
  const float* Wq = (const float*)d_in[4];
  const float* bq = (const float*)d_in[5];
  const float* Wk = (const float*)d_in[6];
  const float* bk = (const float*)d_in[7];
  const float* Wv = (const float*)d_in[8];
  const float* bv = (const float*)d_in[9];
  const float* Wo = (const float*)d_in[10];
  const float* bo = (const float*)d_in[11];
  float* out = (float*)d_out;

  const size_t E  = (size_t)N_TOK * D_MODEL;     // 8,388,608 halves
  const size_t WE = (size_t)D_MODEL * D_MODEL;   // 1,048,576 halves
  _Float16* qh  = (_Float16*)d_ws;
  _Float16* kh  = qh + E;
  _Float16* vt  = kh + E;
  _Float16* x2  = vt + E;
  _Float16* wqh = x2 + E;
  _Float16* wkh = wqh + WE;
  _Float16* wvh = wkh + WE;
  _Float16* woh = wvh + WE;

  const dim3 blk(256);
  const int cvt_blocks  = (int)(WE / (256 * 4));                    // 1024
  const int proj_blocks = (N_TOK / 32) * (D_MODEL / 64) / 8;        // 512
  const int attn_blocks = BATCH * HEADS * (SEQ / 128);              // 1024

  cvt_kernel<<<cvt_blocks, blk, 0, stream>>>(Wq, wqh);
  cvt_kernel<<<cvt_blocks, blk, 0, stream>>>(Wk, wkh);
  cvt_kernel<<<cvt_blocks, blk, 0, stream>>>(Wv, wvh);
  cvt_kernel<<<cvt_blocks, blk, 0, stream>>>(Wo, woh);

  proj_kernel<0><<<proj_blocks, blk, 0, stream>>>(q, wqh, bq, qh);
  proj_kernel<0><<<proj_blocks, blk, 0, stream>>>(k, wkh, bk, kh);
  proj_kernel<1><<<proj_blocks, blk, 0, stream>>>(v, wvh, bv, vt);
  attn_kernel<<<attn_blocks, blk, 0, stream>>>(qh, kh, vt, x2);
  oproj_kernel<<<proj_blocks, blk, 0, stream>>>(x2, woh, bo, out);
}
